// AgentAttention_7155415515542
// MI455X (gfx1250) — matrix operand order repaired and checked by reading, not compiled, not run on hardware
//
#include <hip/hip_runtime.h>

// ---------------------------------------------------------------------------
// Agent attention for MI455X (gfx1250, wave32, WMMA bf16 16x16x32).
// All matmuls run on v_wmma_f32_16x16x32_bf16; the QKV GEMM stages K-slices
// into LDS with the Tensor Data Mover (tensor_load_to_lds, TENSORcnt),
// double-buffered to overlap DMA with WMMA issue.
// ---------------------------------------------------------------------------

typedef unsigned short ushort_t;
typedef __attribute__((ext_vector_type(16))) __bf16        v16bf;
typedef __attribute__((ext_vector_type(8)))  float         v8f;
typedef __attribute__((ext_vector_type(4)))  unsigned int  u32x4;
typedef __attribute__((ext_vector_type(8)))  int           i32x8;
typedef __attribute__((ext_vector_type(4)))  int           i32x4;

#define S_LEN 2048
#define HID   768
#define NH    12
#define HD    64
#define BATCH 16
#define AG    40
#define AGP   48           // agents padded to 3x16 WMMA rows
#define EROWS 4112         // 4095 real rows of dist_emb, padded
#define B1W   2112         // B1 window width (>= 2087), multiple of 16
#define QK_SCALE 0.125f    // 64^-0.5

// ---- bf16 helpers (bit-level, round-to-nearest-even) ----------------------
__device__ inline ushort_t f2bf(float f) {
  unsigned u = __float_as_uint(f);
  unsigned r = u + 0x7fffu + ((u >> 16) & 1u);
  return (ushort_t)(r >> 16);
}
__device__ inline float bf2f(ushort_t h) {
  return __uint_as_float(((unsigned)h) << 16);
}

union FragU { u32x4 q[2]; v16bf v; };

// A fragment (16x32 bf16, M x K): lane = (row, khalf); contiguous 16 elements at K = kh*16
__device__ inline v16bf load_a16(const ushort_t* base, int ld, int lane) {
  int r = lane & 15, kh = lane >> 4;
  const ushort_t* p = base + r * ld + kh * 16;
  FragU f;
  f.q[0] = *(const u32x4*)(p);
  f.q[1] = *(const u32x4*)(p + 8);
  return f.v;
}
// B fragment (32x16 bf16, K x N), source stored [N][K]: lane = (col, khalf);
// contiguous 16 elements at K = kh*16
__device__ inline v16bf load_b16(const ushort_t* base, int ld, int lane) {
  int c = lane & 15, kh = lane >> 4;
  const ushort_t* p = base + c * ld + kh * 16;
  FragU f;
  f.q[0] = *(const u32x4*)(p);
  f.q[1] = *(const u32x4*)(p + 8);
  return f.v;
}
__device__ inline v8f wmma_bf(v16bf a, v16bf b, v8f c) {
  return __builtin_amdgcn_wmma_f32_16x16x32_bf16(false, a, false, b,
                                                 (short)0, c, false, false);
}
__device__ inline v8f v8zero() {
  v8f z = {0.f, 0.f, 0.f, 0.f, 0.f, 0.f, 0.f, 0.f};
  return z;
}

// ---- Tensor Data Mover: 2-D bf16 tile (tile_k x tile_rows) -> LDS ---------
// D# group0: count=1 | lds_addr | 57-bit global addr | type=2
// D# group1: data_size=2B, tensor_dim0/1, tile_dim0/1, dim0 stride
// 6-arg builtin (clang-23 / therock signature): (g0, g1, g2, g3, g4, cpol)
__device__ inline void tdm_load_2d(unsigned lds_off, const ushort_t* gptr,
                                   unsigned rows, unsigned rowlen,
                                   unsigned tile_k, unsigned tile_rows,
                                   unsigned stride) {
  unsigned long long ga = (unsigned long long)(uintptr_t)gptr;
  u32x4 g0;
  g0[0] = 1u;                                           // count=1, user mode
  g0[1] = lds_off;                                      // lds byte address
  g0[2] = (unsigned)(ga & 0xffffffffu);                 // global addr [95:64]
  g0[3] = (unsigned)((ga >> 32) & 0x01ffffffu) | 0x80000000u;  // type=2
  i32x8 g1;
  g1[0] = (int)(1u << 16);                              // data_size = 2 bytes
  g1[1] = (int)((rowlen & 0xffffu) << 16);              // tensor_dim0 lo16
  g1[2] = (int)((rowlen >> 16) | ((rows & 0xffffu) << 16));   // dim0 hi / dim1 lo
  g1[3] = (int)((rows >> 16) | (tile_k << 16));         // dim1 hi / tile_dim0
  g1[4] = (int)tile_rows;                               // tile_dim1 (tile_dim2=0)
  g1[5] = (int)stride;                                  // tensor_dim0_stride lo
  g1[6] = 0;
  g1[7] = 0;
  i32x4 z4 = {0, 0, 0, 0};                              // 2-D: groups 2/3 unused
  i32x8 z8 = {0, 0, 0, 0, 0, 0, 0, 0};
  __builtin_amdgcn_tensor_load_to_lds(g0, g1, z4, z4, z8, 0);
}

// ---- conversions -----------------------------------------------------------
__global__ void k_cvt_x(const float* __restrict__ x, ushort_t* __restrict__ o, int n4) {
  int i = blockIdx.x * blockDim.x + threadIdx.x;
  if (i >= n4) return;
  float4 f = ((const float4*)x)[i];
  uint2 p;
  p.x = (unsigned)f2bf(f.x) | ((unsigned)f2bf(f.y) << 16);
  p.y = (unsigned)f2bf(f.z) | ((unsigned)f2bf(f.w) << 16);
  ((uint2*)o)[i] = p;
}

// Wt[g][n][k] = W_g[k][n]  (bf16, transposed so GEMM B-side is K-contiguous)
__global__ void k_cvt_w(const float* __restrict__ Wq, const float* __restrict__ Wk,
                        const float* __restrict__ Wv, ushort_t* __restrict__ Wt) {
  int id = blockIdx.x * blockDim.x + threadIdx.x;
  if (id >= 3 * HID * HID) return;
  int g = id / (HID * HID);
  int rem = id - g * HID * HID;
  int k = rem / HID, n = rem - (rem / HID) * HID;
  const float* W = (g == 0) ? Wq : ((g == 1) ? Wk : Wv);
  Wt[g * HID * HID + n * HID + k] = f2bf(W[k * HID + n]);
}

__global__ void k_cvt_e(const float* __restrict__ E, ushort_t* __restrict__ Ebf) {
  int id = blockIdx.x * blockDim.x + threadIdx.x;
  if (id >= EROWS * HD) return;
  int r = id >> 6;
  float v = (r < (2 * 2048 - 1)) ? E[id] : 0.f;
  Ebf[id] = f2bf(v);
}

// ---- QKV GEMM: [32768 x 768] x [768 x 768] + bias, bf16 WMMA ---------------
// grid (6, 256, 3), block 256 (8 waves, 2x4 wave grid, 64x32 per wave).
// A/B K-slices (128x32 bf16) staged into LDS by the TDM, double buffered.
__global__ __launch_bounds__(256)
void k_qkv(const ushort_t* __restrict__ Xbf, const ushort_t* __restrict__ Wt,
           const float* __restrict__ bq, const float* __restrict__ bk,
           const float* __restrict__ bv,
           ushort_t* __restrict__ Qbf, ushort_t* __restrict__ Kbf,
           ushort_t* __restrict__ VT) {
  __shared__ ushort_t tA[2][128 * 32];
  __shared__ ushort_t tB[2][128 * 32];
  const int g = blockIdx.z;
  const int tileN = blockIdx.x * 128;
  const int tileM = blockIdx.y * 128;
  const int wave = threadIdx.x >> 5, lane = threadIdx.x & 31;
  const int mw = wave >> 2, nw = wave & 3;
  const ushort_t* Ag = Xbf + tileM * HID;                 // row-major [128+][768]
  const ushort_t* Bg = Wt + g * HID * HID + tileN * HID;  // row-major [128+][768]

  v8f acc[4][2];
  for (int i = 0; i < 4; i++)
    for (int j = 0; j < 2; j++) acc[i][j] = v8zero();

  if (wave == 0) {
    tdm_load_2d((unsigned)(uintptr_t)&tA[0][0], Ag, 32768u, 768u, 32u, 128u, 768u);
    tdm_load_2d((unsigned)(uintptr_t)&tB[0][0], Bg, 768u, 768u, 32u, 128u, 768u);
  }

  for (int it = 0; it < 24; ++it) {
    if (wave == 0) {
      if (it < 23) {
        int k0n = (it + 1) * 32;
        int nb = (it + 1) & 1;
        tdm_load_2d((unsigned)(uintptr_t)&tA[nb][0], Ag + k0n, 32768u, 768u, 32u, 128u, 768u);
        tdm_load_2d((unsigned)(uintptr_t)&tB[nb][0], Bg + k0n, 768u, 768u, 32u, 128u, 768u);
        __builtin_amdgcn_s_wait_tensorcnt(2);   // current buffer complete
      } else {
        __builtin_amdgcn_s_wait_tensorcnt(0);
      }
    }
    __syncthreads();                            // publish LDS tile
    const ushort_t* As = &tA[it & 1][0];
    const ushort_t* Bs = &tB[it & 1][0];
    v16bf a[4], b[2];
    for (int i = 0; i < 4; i++)
      a[i] = load_a16(As + (mw * 64 + i * 16) * 32, 32, lane);
    for (int j = 0; j < 2; j++)
      b[j] = load_b16(Bs + (nw * 32 + j * 16) * 32, 32, lane);
    for (int i = 0; i < 4; i++)
      for (int j = 0; j < 2; j++) acc[i][j] = wmma_bf(a[i], b[j], acc[i][j]);
    __syncthreads();                            // protect buffer vs. overwrite
  }

  const float* bias = (g == 0) ? bq : ((g == 1) ? bk : bv);
  ushort_t* Out = (g == 0) ? Qbf : Kbf;
  const int ln = lane & 15, lh = lane >> 4;
  const int m0 = tileM + mw * 64, n0 = tileN + nw * 32;
  for (int j = 0; j < 2; j++) {
    int col = n0 + j * 16 + ln;
    float bval = bias[col];
    for (int i = 0; i < 4; i++) {
      for (int v = 0; v < 8; v++) {
        int row = m0 + i * 16 + v + 8 * lh;
        float val = acc[i][j][v] + bval;
        if (g < 2) {
          Out[row * HID + col] = f2bf(val);
        } else {
          // store V per-head transposed: VT[(b*12+h)][d][s]
          int b_ = row >> 11, s = row & 2047, h = col >> 6, d = col & 63;
          VT[(b_ * NH + h) * (HD * S_LEN) + d * S_LEN + s] = f2bf(val);
        }
      }
    }
  }
}

// ---- agent pooling (bilinear interp of q along seq to 40 rows, pad to 48) --
__global__ void k_agent(const ushort_t* __restrict__ Qbf, ushort_t* __restrict__ Agt) {
  int id = blockIdx.x * blockDim.x + threadIdx.x;
  if (id >= BATCH * AGP * HID) return;
  int c = id % HID;
  int t = id / HID;
  int a = t % AGP, b = t / AGP;
  float val = 0.f;
  if (a < AG) {
    float src = (a + 0.5f) * ((float)S_LEN / (float)AG) - 0.5f;
    src = fminf(fmaxf(src, 0.f), (float)(S_LEN - 1));
    int lo = (int)floorf(src);
    int hi = lo + 1; if (hi > S_LEN - 1) hi = S_LEN - 1;
    float w = src - (float)lo;
    float qlo = bf2f(Qbf[(b * S_LEN + lo) * HID + c]);
    float qhi = bf2f(Qbf[(b * S_LEN + hi) * HID + c]);
    val = qlo * (1.f - w) + qhi * w;
  }
  Agt[id] = f2bf(val);
}

// ---- B1 = agent @ dist_emb^T over used window (GEMM form of bias1) ---------
// grid (132, 192), block 32 (1 wave). B1[bh][a][j], bias1[a,s] = B1[a, a-s+2047]
__global__ __launch_bounds__(32)
void k_b1(const ushort_t* __restrict__ Agt, const ushort_t* __restrict__ Ebf,
          float* __restrict__ B1) {
  int bh = blockIdx.y;
  int b = bh / NH, h = bh % NH;
  int j0 = blockIdx.x * 16;
  int lane = threadIdx.x;
  const ushort_t* abase = Agt + b * AGP * HID + h * HD;
  v8f acc[3]; for (int i = 0; i < 3; i++) acc[i] = v8zero();
  for (int ks = 0; ks < 2; ks++) {
    v16bf bfr = load_b16(Ebf + j0 * HD + ks * 32, HD, lane);
    for (int i = 0; i < 3; i++) {
      v16bf afr = load_a16(abase + i * 16 * HID + ks * 32, HID, lane);
      acc[i] = wmma_bf(afr, bfr, acc[i]);
    }
  }
  int ln = lane & 15, lh = lane >> 4;
  for (int i = 0; i < 3; i++)
    for (int v = 0; v < 8; v++) {
      int a = i * 16 + v + 8 * lh;
      B1[(bh * AGP + a) * B1W + j0 + ln] = acc[i][v];
    }
}

// ---- stage1 scores: SCALE*agent@k^T + B1 gather + mask ---------------------
// grid (4, 192), block 256 (each wave: 64 s-columns)
__global__ __launch_bounds__(256)
void k_s1score(const ushort_t* __restrict__ Agt, const ushort_t* __restrict__ Kbf,
               const float* __restrict__ B1, const float* __restrict__ mask,
               float* __restrict__ S1) {
  int bh = blockIdx.y;
  int b = bh / NH, h = bh % NH;
  int wave = threadIdx.x >> 5, lane = threadIdx.x & 31;
  int s0 = blockIdx.x * 512 + wave * 64;
  const ushort_t* abase = Agt + b * AGP * HID + h * HD;
  const ushort_t* kbase = Kbf + (b * S_LEN) * HID + h * HD;
  v8f acc[3][4];
  for (int i = 0; i < 3; i++)
    for (int j = 0; j < 4; j++) acc[i][j] = v8zero();
  for (int ks = 0; ks < 2; ks++) {
    v16bf afr[3], bfr[4];
    for (int i = 0; i < 3; i++)
      afr[i] = load_a16(abase + i * 16 * HID + ks * 32, HID, lane);
    for (int j = 0; j < 4; j++)
      bfr[j] = load_b16(kbase + (s0 + j * 16) * HID + ks * 32, HID, lane);
    for (int i = 0; i < 3; i++)
      for (int j = 0; j < 4; j++) acc[i][j] = wmma_bf(afr[i], bfr[j], acc[i][j]);
  }
  int ln = lane & 15, lh = lane >> 4;
  for (int i = 0; i < 3; i++)
    for (int j = 0; j < 4; j++) {
      int scol = s0 + j * 16 + ln;
      float mval = mask[b * S_LEN + scol];
      for (int v = 0; v < 8; v++) {
        int a = i * 16 + v + 8 * lh;
        float sc = QK_SCALE * acc[i][j][v]
                 + B1[(bh * AGP + a) * B1W + (a + 2047 - scol)] + mval;
        S1[(bh * AGP + a) * S_LEN + scol] = sc;
      }
    }
}

// ---- stage1 softmax rows -> P (bf16), rows >= 40 zeroed (K-dim padding) ----
__global__ __launch_bounds__(256)
void k_s1soft(const float* __restrict__ S1, ushort_t* __restrict__ P) {
  int row = blockIdx.x;            // bh*48 + a
  int a = row % AGP;
  int t = threadIdx.x;
  __shared__ float red[256];
  const float* src = S1 + row * S_LEN;
  ushort_t* dst = P + row * S_LEN;
  if (a >= AG) {
    for (int i = t; i < S_LEN; i += 256) dst[i] = 0;
    return;
  }
  float vals[8];
  float m = -1e30f;
  for (int i = 0; i < 8; i++) { vals[i] = src[t + i * 256]; m = fmaxf(m, vals[i]); }
  red[t] = m; __syncthreads();
  for (int o = 128; o > 0; o >>= 1) { if (t < o) red[t] = fmaxf(red[t], red[t + o]); __syncthreads(); }
  m = red[0]; __syncthreads();
  float sum = 0.f;
  for (int i = 0; i < 8; i++) { vals[i] = __expf(vals[i] - m); sum += vals[i]; }
  red[t] = sum; __syncthreads();
  for (int o = 128; o > 0; o >>= 1) { if (t < o) red[t] += red[t + o]; __syncthreads(); }
  float inv = 1.f / red[0];
  for (int i = 0; i < 8; i++) dst[t + i * 256] = f2bf(vals[i] * inv);
}

// ---- stage1 agent_v = P @ V (K=2048 split over waves, deterministic) -------
// grid (192), block 256; output AvT[bh][d][a] bf16 (a padded to 64)
__global__ __launch_bounds__(256)
void k_s1av(const ushort_t* __restrict__ P, const ushort_t* __restrict__ VT,
            ushort_t* __restrict__ AvT) {
  int bh = blockIdx.x;
  int wave = threadIdx.x >> 5, lane = threadIdx.x & 31;
  int kw = wave >> 1, nw = wave & 1;          // 4 K-chunks x 2 N-halves
  __shared__ float part[4][AGP][64];
  const ushort_t* pbase = P + bh * AGP * S_LEN;
  const ushort_t* vbase = VT + bh * (HD * S_LEN);
  v8f acc[3][2];
  for (int i = 0; i < 3; i++)
    for (int j = 0; j < 2; j++) acc[i][j] = v8zero();
  int kb = kw * 512;
  for (int ks = 0; ks < 16; ks++) {
    int k0 = kb + ks * 32;
    v16bf afr[3], bfr[2];
    for (int i = 0; i < 3; i++)
      afr[i] = load_a16(pbase + i * 16 * S_LEN + k0, S_LEN, lane);
    for (int j = 0; j < 2; j++)
      bfr[j] = load_b16(vbase + (nw * 32 + j * 16) * S_LEN + k0, S_LEN, lane);
    for (int i = 0; i < 3; i++)
      for (int j = 0; j < 2; j++) acc[i][j] = wmma_bf(afr[i], bfr[j], acc[i][j]);
  }
  int ln = lane & 15, lh = lane >> 4;
  for (int i = 0; i < 3; i++)
    for (int j = 0; j < 2; j++)
      for (int v = 0; v < 8; v++) {
        int a = i * 16 + v + 8 * lh;
        int d = nw * 32 + j * 16 + ln;
        part[kw][a][d] = acc[i][j][v];
      }
  __syncthreads();
  for (int idx = threadIdx.x; idx < 64 * 64; idx += 256) {
    int a = idx & 63, d = idx >> 6;
    float v = 0.f;
    if (a < AGP) v = part[0][a][d] + part[1][a][d] + part[2][a][d] + part[3][a][d];
    AvT[bh * 4096 + d * 64 + a] = f2bf(v);
  }
}

// ---- stage2 fused: score2 + bias2 tile + softmax + x = P2 @ agent_v --------
// grid (128, 192), block 32 (1 wave per 16-row s-tile)
__global__ __launch_bounds__(32)
void k_s2(const ushort_t* __restrict__ Qbf, const ushort_t* __restrict__ Agt,
          const ushort_t* __restrict__ Ebf, const ushort_t* __restrict__ AvT,
          float* __restrict__ Out) {
  int bh = blockIdx.y;
  int b = bh / NH, h = bh % NH;
  int s0 = blockIdx.x * 16;
  int lane = threadIdx.x;
  __shared__ float sc[16][48];
  __shared__ float bt[16][64];
  __shared__ ushort_t pl[16][64];
  const ushort_t* qbase = Qbf + (b * S_LEN + s0) * HID + h * HD;
  const ushort_t* abase = Agt + b * AGP * HID + h * HD;
  int jbase = s0 + 2008;        // bias2 band window: j = s - a + 2047

  v8f accS[3], accB[4];
  for (int i = 0; i < 3; i++) accS[i] = v8zero();
  for (int j = 0; j < 4; j++) accB[j] = v8zero();
  for (int ks = 0; ks < 2; ks++) {
    v16bf qf = load_a16(qbase + ks * 32, HID, lane);
    for (int i = 0; i < 3; i++) {
      v16bf af = load_b16(abase + i * 16 * HID + ks * 32, HID, lane);
      accS[i] = wmma_bf(qf, af, accS[i]);
    }
    for (int j = 0; j < 4; j++) {
      v16bf ef = load_b16(Ebf + (jbase + j * 16) * HD + ks * 32, HD, lane);
      accB[j] = wmma_bf(qf, ef, accB[j]);
    }
  }
  int ln = lane & 15, lh = lane >> 4;
  for (int i = 0; i < 3; i++)
    for (int v = 0; v < 8; v++) sc[v + 8 * lh][i * 16 + ln] = accS[i][v];
  for (int j = 0; j < 4; j++)
    for (int v = 0; v < 8; v++) bt[v + 8 * lh][j * 16 + ln] = accB[j][v];
  __syncthreads();
  if (lane < 16) {
    int r = lane;
    float row[AG];
    float m = -1e30f;
    for (int a = 0; a < AG; a++) {
      float v = QK_SCALE * sc[r][a] + bt[r][r + 39 - a];
      row[a] = v; m = fmaxf(m, v);
    }
    float sum = 0.f;
    for (int a = 0; a < AG; a++) { row[a] = __expf(row[a] - m); sum += row[a]; }
    float inv = 1.f / sum;
    for (int a = 0; a < AG; a++) pl[r][a] = f2bf(row[a] * inv);
    for (int a = AG; a < 64; a++) pl[r][a] = 0;
  }
  __syncthreads();
  v8f accX[4];
  for (int j = 0; j < 4; j++) accX[j] = v8zero();
  for (int ks = 0; ks < 2; ks++) {
    v16bf pf = load_a16(&pl[0][0] + ks * 32, 64, lane);
    for (int j = 0; j < 4; j++) {
      v16bf avf = load_b16(AvT + bh * 4096 + (j * 16) * 64 + ks * 32, 64, lane);
      accX[j] = wmma_bf(pf, avf, accX[j]);
    }
  }
  for (int j = 0; j < 4; j++)
    for (int v = 0; v < 8; v++) {
      int r = v + 8 * lh, d = j * 16 + ln;
      Out[(b * S_LEN + s0 + r) * HID + h * HD + d] = accX[j][v];
    }
}

// ---------------------------------------------------------------------------
extern "C" void kernel_launch(void* const* d_in, const int* in_sizes, int n_in,
                              void* d_out, int out_size, void* d_ws, size_t ws_size,
                              hipStream_t stream) {
  const float* X    = (const float*)d_in[0];
  const float* mask = (const float*)d_in[1];
  const float* Wq   = (const float*)d_in[2];
  const float* bq   = (const float*)d_in[3];
  const float* Wk   = (const float*)d_in[4];
  const float* bk   = (const float*)d_in[5];
  const float* Wv   = (const float*)d_in[6];
  const float* bv   = (const float*)d_in[7];
  const float* E    = (const float*)d_in[8];
  float* Out = (float*)d_out;

  char* w = (char*)d_ws;
  size_t off = 0;
  auto alloc = [&](size_t bytes) -> char* {
    char* p = w + off;
    off += (bytes + 255) & ~(size_t)255;
    return p;
  };
  ushort_t* Xbf  = (ushort_t*)alloc((size_t)BATCH * S_LEN * HID * 2);
  ushort_t* Wt   = (ushort_t*)alloc((size_t)3 * HID * HID * 2);
  ushort_t* Ebf  = (ushort_t*)alloc((size_t)EROWS * HD * 2);
  ushort_t* Qbf  = (ushort_t*)alloc((size_t)BATCH * S_LEN * HID * 2);
  ushort_t* Kbf  = (ushort_t*)alloc((size_t)BATCH * S_LEN * HID * 2);
  ushort_t* VT   = (ushort_t*)alloc((size_t)BATCH * NH * HD * S_LEN * 2);
  ushort_t* Agt  = (ushort_t*)alloc((size_t)BATCH * AGP * HID * 2);
  float*    B1   = (float*)   alloc((size_t)BATCH * NH * AGP * B1W * 4);
  float*    S1   = (float*)   alloc((size_t)BATCH * NH * AGP * S_LEN * 4);
  ushort_t* P    = (ushort_t*)alloc((size_t)BATCH * NH * AGP * S_LEN * 2);
  ushort_t* AvT  = (ushort_t*)alloc((size_t)BATCH * NH * HD * 64 * 2);
  (void)ws_size; (void)in_sizes; (void)n_in; (void)out_size;

  int n4 = BATCH * S_LEN * HID / 4;
  k_cvt_x<<<dim3(n4 / 256), dim3(256), 0, stream>>>(X, Xbf, n4);
  k_cvt_w<<<dim3((3 * HID * HID) / 256), dim3(256), 0, stream>>>(Wq, Wk, Wv, Wt);
  k_cvt_e<<<dim3((EROWS * HD) / 256), dim3(256), 0, stream>>>(E, Ebf);

  k_qkv<<<dim3(6, 256, 3), dim3(256), 0, stream>>>(Xbf, Wt, bq, bk, bv, Qbf, Kbf, VT);

  k_agent<<<dim3((BATCH * AGP * HID) / 256), dim3(256), 0, stream>>>(Qbf, Agt);

  k_b1<<<dim3(B1W / 16, BATCH * NH), dim3(32), 0, stream>>>(Agt, Ebf, B1);

  k_s1score<<<dim3(4, BATCH * NH), dim3(256), 0, stream>>>(Agt, Kbf, B1, mask, S1);
  k_s1soft<<<dim3(BATCH * NH * AGP), dim3(256), 0, stream>>>(S1, P);
  k_s1av<<<dim3(BATCH * NH), dim3(256), 0, stream>>>(P, VT, AvT);

  k_s2<<<dim3(S_LEN / 16, BATCH * NH), dim3(32), 0, stream>>>(Qbf, Agt, Ebf, AvT, Out);
}
